// QuantumTransformer_51049981281070
// MI455X (gfx1250) — compile-verified, matrix operand order checked by reading, not run on hardware
//
#include <hip/hip_runtime.h>
#include <hip/hip_bf16.h>

typedef __attribute__((ext_vector_type(16))) __bf16 v16bf;
typedef __attribute__((ext_vector_type(8)))  __bf16 v8bf;
typedef __attribute__((ext_vector_type(8)))  float  v8f;
typedef __attribute__((ext_vector_type(4)))  unsigned int u32x4;
typedef __attribute__((ext_vector_type(8)))  int i32x8;
typedef __attribute__((ext_vector_type(4)))  int i32x4;

// Problem constants
#define CB   4
#define CS   2048
#define CD   512
#define CH   8
#define CDK  64
#define CDFF 2048
#define CM   (CB*CS)   // 8192 rows

// ---------------------------------------------------------------------------
// Tensor Data Mover availability
// ---------------------------------------------------------------------------
#if defined(__has_builtin)
#if __has_builtin(__builtin_amdgcn_tensor_load_to_lds) && __has_builtin(__builtin_amdgcn_s_wait_tensorcnt)
#define QT_USE_TDM 1
#endif
#endif
#ifndef QT_USE_TDM
#define QT_USE_TDM 0
#endif

#if QT_USE_TDM
// Issue one 2D TDM tile load (bf16 elements, data_size=2B) into LDS.
// D# per CDNA5 ISA:
//   group0: count=1 | lds_addr[63:32] | global_addr[120:64] | type=2 [127:126]
//   group1: data_size=1(2B), pad fields, tensor_dim0/1, tile_dim0/1, dim0_stride
// pad_interval code 3 = 16 DWORDs (64B = one 32-elem bf16 row),
// pad_amount  code 3 = 4 DWORDs (16B = 8 bf16) -> LDS row stride = BK+8 elems.
__device__ __forceinline__ void qt_tdm_load_2d(unsigned lds_addr, unsigned long long gaddr,
                                               unsigned long long dim0, unsigned long long dim1,
                                               unsigned tile0, unsigned tile1,
                                               unsigned long long stride0,
                                               unsigned pad_en, unsigned pad_int, unsigned pad_amt) {
  u32x4 g0;
  g0[0] = 1u;                                                  // count=1 (valid user D#)
  g0[1] = lds_addr;                                            // lds_addr [63:32]
  g0[2] = (unsigned)(gaddr & 0xffffffffull);                   // global_addr [95:64]
  g0[3] = (unsigned)((gaddr >> 32) & 0x01ffffffull) | (2u << 30); // addr[56:32] | type=2
  i32x8 g1;
  g1[0] = (int)((1u << 16) | (pad_en << 20) | (pad_int << 22) | (pad_amt << 25)); // data_size=1(2B)
  g1[1] = (int)((unsigned)(dim0 & 0xffffull) << 16);           // tensor_dim0[15:0] @ bits 63:48
  g1[2] = (int)(((unsigned)(dim0 >> 16) & 0xffffu) | (((unsigned)(dim1 & 0xffffull)) << 16));
  g1[3] = (int)((((unsigned)(dim1 >> 16)) & 0xffffu) | ((tile0 & 0xffffu) << 16));
  g1[4] = (int)(tile1 & 0xffffu);                              // tile_dim1; tile_dim2=0
  g1[5] = (int)(stride0 & 0xffffffffull);                      // dim0_stride[31:0]
  g1[6] = (int)((unsigned)((stride0 >> 32) & 0xffffull));      // dim0_stride[47:32]; dim1_stride=0
  g1[7] = 0;
  i32x4 z4 = {0, 0, 0, 0};
#if __clang_major__ >= 23
  i32x8 z8 = {0, 0, 0, 0, 0, 0, 0, 0};
  __builtin_amdgcn_tensor_load_to_lds(g0, g1, z4, z4, z8, 0);
#else
  __builtin_amdgcn_tensor_load_to_lds(g0, g1, z4, z4, 0);
#endif
}
#endif

// ---------------------------------------------------------------------------
// fp32 -> bf16 (row-major copy, for A-side activations)
// ---------------------------------------------------------------------------
__global__ void qt_cvt_bf16(const float* __restrict__ src, __bf16* __restrict__ dst, int n) {
  int i = blockIdx.x * blockDim.x + threadIdx.x;
  if (i < n) dst[i] = (__bf16)src[i];
}

// fp32 (K x N) -> bf16 transposed (N x K) for B-side weights
__global__ void qt_cvt_transpose_bf16(const float* __restrict__ src, __bf16* __restrict__ dst,
                                      int K, int N) {
  int i = blockIdx.x * blockDim.x + threadIdx.x;
  if (i >= K * N) return;
  int n = i / K, k = i - n * K;
  dst[i] = (__bf16)src[(size_t)k * N + n];
}

// ---------------------------------------------------------------------------
// Pack the 4 used q/k columns per head into transposed 128 x 512 bf16 weights
// (rows 0..31 = q-features, 32..63 = k-features, 64..127 zero), + packed bias.
// ---------------------------------------------------------------------------
__global__ void qt_pack_qk(const float* __restrict__ wq, const float* __restrict__ wk,
                           const float* __restrict__ bq, const float* __restrict__ bk,
                           __bf16* __restrict__ wqkbT, float* __restrict__ bqk) {
  int i = blockIdx.x * blockDim.x + threadIdx.x;   // over 128*512
  if (i >= 128 * CD) return;
  int c = i / CD, kk = i - c * CD;
  float v = 0.f;
  if (c < 32)      { int h = c >> 2, n = c & 3;        v = wq[(size_t)kk * CD + h * CDK + n]; }
  else if (c < 64) { int h = (c - 32) >> 2, n = c & 3; v = wk[(size_t)kk * CD + h * CDK + n]; }
  wqkbT[i] = (__bf16)v;
  if (kk == 0) {
    float bb = 0.f;
    if (c < 32)      { int h = c >> 2, n = c & 3;        bb = bq[h * CDK + n]; }
    else if (c < 64) { int h = (c - 32) >> 2, n = c & 3; bb = bk[h * CDK + n]; }
    bqk[c] = bb;
  }
}

// ---------------------------------------------------------------------------
// bf16 WMMA GEMM: C = A(MxK) * Bt(NxK)^T + bias, optional relu/residual.
// Block tile 128x128, BK=32, 256 threads = 8 waves (4x2), wave tile 32x64.
// Tiles staged into LDS by the Tensor Data Mover (double buffered).
// mode: 0 = fp32 row-major, 1 = bf16 row-major, 2 = bf16 (B,H,DK,S) v^T layout
// Row count per batch is always CS (compile-time) -> shift/mask addressing.
// ---------------------------------------------------------------------------
__global__ __launch_bounds__(256)
void qt_gemm_bf16(const __bf16* __restrict__ A, const __bf16* __restrict__ Bt,
                  const float* __restrict__ bias, const float* __restrict__ res,
                  float* __restrict__ outF, __bf16* __restrict__ outB,
                  int M, int N, int K, int mode, int do_relu) {
  constexpr int BM = 128, BN = 128, BK = 32, PAD = 8;
  __shared__ __bf16 lA[2][BM][BK + PAD];   // [m][k], row stride 80B
  __shared__ __bf16 lB[2][BN][BK + PAD];   // [n][k], row stride 80B

  const int t = threadIdx.x;
  const int w = t >> 5, lane = t & 31;
  const int m = lane & 15, half = lane >> 4;
  const int wm = w >> 1, wn = w & 1;
  const int m0 = blockIdx.y * BM, n0 = blockIdx.x * BN;

  v8f acc[2][4];
  for (int i = 0; i < 2; i++)
    for (int j = 0; j < 4; j++)
      for (int r = 0; r < 8; r++) acc[i][j][r] = 0.f;

  const int nk = K / BK;

#if QT_USE_TDM
  auto issue = [&](int kt, int buf) {
    int k0 = kt * BK;
    qt_tdm_load_2d((unsigned)(size_t)&lA[buf][0][0],
                   (unsigned long long)(size_t)(A + (size_t)m0 * K + k0),
                   (unsigned long long)K, (unsigned long long)M, BK, BM,
                   (unsigned long long)K, 1u, 3u, 3u);
    qt_tdm_load_2d((unsigned)(size_t)&lB[buf][0][0],
                   (unsigned long long)(size_t)(Bt + (size_t)n0 * K + k0),
                   (unsigned long long)K, (unsigned long long)N, BK, BN,
                   (unsigned long long)K, 1u, 3u, 3u);
  };
  if (w == 0) {
    issue(0, 0);
    __builtin_amdgcn_s_wait_tensorcnt(0);
  }
  __syncthreads();
#else
  auto stage = [&](int kt, int buf) {
    int k0 = kt * BK;
    for (int i = 0; i < 2; i++) {               // A: 128 rows x 4 16B-chunks
      int c = t + 256 * i;
      int row = c >> 2, cc = c & 3;
      *(uint4*)&lA[buf][row][cc * 8] = *(const uint4*)(A + (size_t)(m0 + row) * K + k0 + cc * 8);
    }
    for (int i = 0; i < 2; i++) {               // Bt: 128 rows x 4 16B-chunks
      int c = t + 256 * i;
      int row = c >> 2, cc = c & 3;
      *(uint4*)&lB[buf][row][cc * 8] = *(const uint4*)(Bt + (size_t)(n0 + row) * K + k0 + cc * 8);
    }
  };
  stage(0, 0);
  __syncthreads();
#endif

  for (int kt = 0; kt < nk; kt++) {
    int buf = kt & 1;
#if QT_USE_TDM
    if (kt + 1 < nk && w == 0) issue(kt + 1, buf ^ 1);
#endif

    v16bf af[2];
    for (int i = 0; i < 2; i++) {
      const __bf16* base = &lA[buf][wm * 32 + i * 16 + m][half * 8];
      v8bf lo = *(const v8bf*)base;
      v8bf hi = *(const v8bf*)(base + 16);
      for (int e = 0; e < 8; e++) { af[i][e] = lo[e]; af[i][e + 8] = hi[e]; }
    }
    v16bf bfr[4];
    for (int j = 0; j < 4; j++) {
      const __bf16* base = &lB[buf][wn * 64 + j * 16 + m][half * 8];
      v8bf lo = *(const v8bf*)base;
      v8bf hi = *(const v8bf*)(base + 16);
      for (int e = 0; e < 8; e++) { bfr[j][e] = lo[e]; bfr[j][e + 8] = hi[e]; }
    }
    for (int i = 0; i < 2; i++)
      for (int j = 0; j < 4; j++)
        acc[i][j] = __builtin_amdgcn_wmma_f32_16x16x32_bf16(
            false, af[i], false, bfr[j], (short)0, acc[i][j], false, false);

    if (kt + 1 < nk) {
#if QT_USE_TDM
      if (w == 0) __builtin_amdgcn_s_wait_tensorcnt(0);
      __syncthreads();
#else
      __syncthreads();
      stage(kt + 1, buf ^ 1);
      __syncthreads();
#endif
    }
  }

  // Epilogue. C layout: vgpr r, lane l -> M = r + 8*(l>>4), N = l&15.
  for (int i = 0; i < 2; i++)
    for (int j = 0; j < 4; j++) {
      int gcol = n0 + wn * 64 + j * 16 + m;
      for (int r = 0; r < 8; r++) {
        int grow = m0 + wm * 32 + i * 16 + half * 8 + r;
        float v = acc[i][j][r] + bias[gcol];
        if (do_relu) v = fmaxf(v, 0.f);
        if (res) v += res[(size_t)grow * N + gcol];
        if (mode == 0) {
          outF[(size_t)grow * N + gcol] = v;
        } else if (mode == 1) {
          outB[(size_t)grow * N + gcol] = (__bf16)v;
        } else {  // V^T heads layout: (B, H, DK, S); S == CS (compile-time)
          int b = grow >> 11, s = grow & (CS - 1);
          int h = gcol >> 6, d = gcol & 63;
          outB[(((size_t)(b * CH + h)) * CDK + d) * CS + s] = (__bf16)v;
        }
      }
    }
}

// ---------------------------------------------------------------------------
// Rank-5 attention features f=[q0..q3, sum] per (b,h,s), padded to 8 floats.
// ---------------------------------------------------------------------------
__global__ void qt_build_features(const float* __restrict__ qk,
                                  float* __restrict__ fqf, float* __restrict__ fkf) {
  int i = blockIdx.x * blockDim.x + threadIdx.x;     // over 8192*8
  if (i >= CM * CH) return;
  int row = i >> 3, h = i & 7;
  int b = row >> 11, s = row & (CS - 1);
  size_t ob = (((size_t)(b * CH + h)) * CS + s) * 8;
  float q0 = qk[row * 128 + h * 4 + 0], q1 = qk[row * 128 + h * 4 + 1];
  float q2 = qk[row * 128 + h * 4 + 2], q3 = qk[row * 128 + h * 4 + 3];
  fqf[ob + 0] = q0; fqf[ob + 1] = q1; fqf[ob + 2] = q2; fqf[ob + 3] = q3;
  fqf[ob + 4] = q0 + q1 + q2 + q3; fqf[ob + 5] = 0.f; fqf[ob + 6] = 0.f; fqf[ob + 7] = 0.f;
  float k0 = qk[row * 128 + 32 + h * 4 + 0], k1 = qk[row * 128 + 32 + h * 4 + 1];
  float k2 = qk[row * 128 + 32 + h * 4 + 2], k3 = qk[row * 128 + 32 + h * 4 + 3];
  fkf[ob + 0] = k0; fkf[ob + 1] = k1; fkf[ob + 2] = k2; fkf[ob + 3] = k3;
  fkf[ob + 4] = k0 + k1 + k2 + k3; fkf[ob + 5] = 0.f; fkf[ob + 6] = 0.f; fkf[ob + 7] = 0.f;
}

// ---------------------------------------------------------------------------
// Flash attention with rank-5 scores.  Grid (B*H, S/128), block 256 = 8 waves.
// Each wave owns 16 query rows; keys streamed 32 at a time.  V^T tiles and key
// features are double buffered: next tile's TDM + feature stores are issued
// before the current tile's compute, one barrier per iteration.
// ---------------------------------------------------------------------------
__global__ __launch_bounds__(256)
void qt_flash_attn(const float* __restrict__ fqf, const float* __restrict__ fkf,
                   const __bf16* __restrict__ vbT, __bf16* __restrict__ ob) {
  constexpr int KT = 32;
  __shared__ float  lFK[2][KT][8];
  __shared__ __bf16 lV[2][CDK][KT + 8];     // [dk][key], row stride 80B

  const int t = threadIdx.x, w = t >> 5, lane = t & 31;
  const int m = lane & 15, half = lane >> 4;
  const int bh = blockIdx.x;                       // b*H + h
  const int q0 = blockIdx.y * 128 + w * 16;

  const float* fq = fqf + (((size_t)bh) * CS + q0 + m) * 8;
  float f0 = fq[0], f1 = fq[1], f2 = fq[2], f3 = fq[3], f4 = fq[4];

  v8f acc[4];
  for (int j = 0; j < 4; j++)
    for (int r = 0; r < 8; r++) acc[j][r] = 0.f;
  float mrun = -1e30f, lrun = 0.f;

  const __bf16* vbase = vbT + ((size_t)bh) * CDK * CS;   // 64 x 2048 matrix
  const float*  kbase = fkf + ((size_t)bh) * CS * 8;
  const int nkt = CS / KT;

  auto stage = [&](int kt, int buf) {
    int k0 = kt * KT;
#if QT_USE_TDM
    if (w == 0)
      qt_tdm_load_2d((unsigned)(size_t)&lV[buf][0][0],
                     (unsigned long long)(size_t)(vbase + k0),
                     (unsigned long long)CS, (unsigned long long)CDK, KT, CDK,
                     (unsigned long long)CS, 1u, 3u, 3u);
#else
    {   // V^T tile: 64 rows x 2 16B-chunks each -> straight copies
      int dk = t >> 2, cc = t & 3;
      if (cc < 2)
        *(uint4*)&lV[buf][dk][cc * 8] = *(const uint4*)(vbase + (size_t)dk * CS + k0 + cc * 8);
    }
#endif
    {   // key features: 32 keys x 8 floats, 1 per thread
      int key = t >> 3, f = t & 7;
      lFK[buf][key][f] = kbase[(size_t)(k0 + key) * 8 + f];
    }
  };

  stage(0, 0);
#if QT_USE_TDM
  if (w == 0) __builtin_amdgcn_s_wait_tensorcnt(0);
#endif
  __syncthreads();

  for (int kt = 0; kt < nkt; kt++) {
    int buf = kt & 1;
    if (kt + 1 < nkt) stage(kt + 1, buf ^ 1);

    // scores for this lane's 16 key slots (A-fragment key ordering)
    float sc16[16];
    float tmax = -1e30f;
    for (int jj = 0; jj < 16; jj++) {
      int kl = (jj < 8) ? (half * 8 + jj) : (16 + half * 8 + (jj - 8));
      float sv = 0.25f * (f0 * lFK[buf][kl][0] + f1 * lFK[buf][kl][1] + f2 * lFK[buf][kl][2] +
                          f3 * lFK[buf][kl][3] + f4 * lFK[buf][kl][4]);
      sc16[jj] = sv;
      tmax = fmaxf(tmax, sv);
    }
    tmax = fmaxf(tmax, __shfl_xor(tmax, 16, 32));
    float mnew  = fmaxf(mrun, tmax);
    float scale = __expf(mrun - mnew);
    float psum  = 0.f;
    v16bf pf;
    for (int jj = 0; jj < 16; jj++) {
      float p = __expf(sc16[jj] - mnew);
      psum += p;
      pf[jj] = (__bf16)p;
    }
    psum += __shfl_xor(psum, 16, 32);
    lrun = lrun * scale + psum;
    mrun = mnew;

    // rescale accumulators: row of element r is 8*half + r
    float scr[8];
    for (int r = 0; r < 8; r++) scr[r] = __shfl(scale, half * 8 + r, 32);
    for (int j = 0; j < 4; j++)
      for (int r = 0; r < 8; r++) acc[j][r] *= scr[r];

    for (int j = 0; j < 4; j++) {
      const __bf16* base = &lV[buf][j * 16 + m][half * 8];
      v8bf lo = *(const v8bf*)base;
      v8bf hi = *(const v8bf*)(base + 16);
      v16bf vf;
      for (int e = 0; e < 8; e++) { vf[e] = lo[e]; vf[e + 8] = hi[e]; }
      acc[j] = __builtin_amdgcn_wmma_f32_16x16x32_bf16(
          false, pf, false, vf, (short)0, acc[j], false, false);
    }

    if (kt + 1 < nkt) {
#if QT_USE_TDM
      if (w == 0) __builtin_amdgcn_s_wait_tensorcnt(0);
#endif
      __syncthreads();
    }
  }

  float inv = 1.f / lrun;
  float invr[8];
  for (int r = 0; r < 8; r++) invr[r] = __shfl(inv, half * 8 + r, 32);
  int b = bh >> 3, h = bh & 7;
  for (int j = 0; j < 4; j++) {
    int col = h * CDK + j * 16 + m;
    for (int r = 0; r < 8; r++) {
      int srow = q0 + half * 8 + r;
      ob[(((size_t)b) * CS + srow) * CD + col] = (__bf16)(acc[j][r] * invr[r]);
    }
  }
}

// ---------------------------------------------------------------------------
// LayerNorm over D=512, one row per block (256 threads, 2 elems each).
// ---------------------------------------------------------------------------
__global__ __launch_bounds__(256)
void qt_layernorm(const float* __restrict__ x, const float* __restrict__ g,
                  const float* __restrict__ be, float* __restrict__ outF,
                  __bf16* __restrict__ outB) {
  __shared__ float red[256];
  int row = blockIdx.x, t = threadIdx.x;
  const float* xr = x + (size_t)row * CD;
  float a = xr[t], b = xr[t + 256];
  red[t] = a + b;
  __syncthreads();
  for (int o = 128; o > 0; o >>= 1) { if (t < o) red[t] += red[t + o]; __syncthreads(); }
  float mean = red[0] * (1.f / CD);
  __syncthreads();
  float da = a - mean, db = b - mean;
  red[t] = da * da + db * db;
  __syncthreads();
  for (int o = 128; o > 0; o >>= 1) { if (t < o) red[t] += red[t + o]; __syncthreads(); }
  float rstd = rsqrtf(red[0] * (1.f / CD) + 1e-5f);
  float ya = da * rstd * g[t] + be[t];
  float yb = db * rstd * g[t + 256] + be[t + 256];
  outF[(size_t)row * CD + t] = ya;
  outF[(size_t)row * CD + t + 256] = yb;
  if (outB) {
    outB[(size_t)row * CD + t] = (__bf16)ya;
    outB[(size_t)row * CD + t + 256] = (__bf16)yb;
  }
}

// ---------------------------------------------------------------------------
extern "C" void kernel_launch(void* const* d_in, const int* in_sizes, int n_in,
                              void* d_out, int out_size, void* d_ws, size_t ws_size,
                              hipStream_t stream) {
  const float* x   = (const float*)d_in[0];
  const float* wq  = (const float*)d_in[1];
  const float* bq  = (const float*)d_in[2];
  const float* wk  = (const float*)d_in[3];
  const float* bk  = (const float*)d_in[4];
  const float* wv  = (const float*)d_in[5];
  const float* bv  = (const float*)d_in[6];
  const float* wo  = (const float*)d_in[7];
  const float* bo  = (const float*)d_in[8];
  const float* w1  = (const float*)d_in[9];
  const float* b1  = (const float*)d_in[10];
  const float* w2  = (const float*)d_in[11];
  const float* b2  = (const float*)d_in[12];
  const float* g1  = (const float*)d_in[13];
  const float* be1 = (const float*)d_in[14];
  const float* g2  = (const float*)d_in[15];
  const float* be2 = (const float*)d_in[16];

  char* ws = (char*)d_ws;
  size_t off = 0;
  auto alloc = [&](size_t bytes) -> void* {
    void* p = ws + off;
    off = (off + bytes + 255) & ~(size_t)255;
    return p;
  };
  __bf16* xb    = (__bf16*)alloc((size_t)CM * CD * 2);
  __bf16* wvbT  = (__bf16*)alloc((size_t)CD * CD * 2);
  __bf16* wobT  = (__bf16*)alloc((size_t)CD * CD * 2);
  __bf16* w1bT  = (__bf16*)alloc((size_t)CD * CDFF * 2);
  __bf16* w2bT  = (__bf16*)alloc((size_t)CDFF * CD * 2);
  __bf16* wqkbT = (__bf16*)alloc((size_t)128 * CD * 2);
  float*  bqk   = (float*)alloc(128 * 4);
  float*  qk    = (float*)alloc((size_t)CM * 128 * 4);
  float*  fqf   = (float*)alloc((size_t)CB * CH * CS * 8 * 4);
  float*  fkf   = (float*)alloc((size_t)CB * CH * CS * 8 * 4);
  __bf16* vbufT = (__bf16*)alloc((size_t)CM * CD * 2);
  __bf16* obuf  = (__bf16*)alloc((size_t)CM * CD * 2);
  float*  tbuf  = (float*)alloc((size_t)CM * CD * 4);
  float*  x1    = (float*)alloc((size_t)CM * CD * 4);
  __bf16* x1b   = (__bf16*)alloc((size_t)CM * CD * 2);
  __bf16* hb    = (__bf16*)alloc((size_t)CM * CDFF * 2);
  (void)ws_size; (void)in_sizes; (void)n_in; (void)out_size;

  // 1) bf16 conversions (A row-major; B transposed to NxK)
  qt_cvt_bf16<<<(CM * CD) / 256, 256, 0, stream>>>(x, xb, CM * CD);
  qt_cvt_transpose_bf16<<<(CD * CD) / 256, 256, 0, stream>>>(wv, wvbT, CD, CD);
  qt_cvt_transpose_bf16<<<(CD * CD) / 256, 256, 0, stream>>>(wo, wobT, CD, CD);
  qt_cvt_transpose_bf16<<<(CD * CDFF) / 256, 256, 0, stream>>>(w1, w1bT, CD, CDFF);
  qt_cvt_transpose_bf16<<<(CDFF * CD) / 256, 256, 0, stream>>>(w2, w2bT, CDFF, CD);
  qt_pack_qk<<<(128 * CD) / 256, 256, 0, stream>>>(wq, wk, bq, bk, wqkbT, bqk);

  // 2) QK feature GEMM (8192 x 128 x 512) -> qk fp32; build rank-5 features
  qt_gemm_bf16<<<dim3(1, CM / 128), 256, 0, stream>>>(
      xb, wqkbT, bqk, nullptr, qk, nullptr, CM, 128, CD, 0, 0);
  qt_build_features<<<(CM * CH) / 256, 256, 0, stream>>>(qk, fqf, fkf);

  // 3) V projection (8192 x 512 x 512), stored bf16 transposed (B,H,DK,S)
  qt_gemm_bf16<<<dim3(CD / 128, CM / 128), 256, 0, stream>>>(
      xb, wvbT, bv, nullptr, nullptr, vbufT, CM, CD, CD, 2, 0);

  // 4) flash attention -> obuf bf16 (B,S,D)
  qt_flash_attn<<<dim3(CB * CH, CS / 128), 256, 0, stream>>>(fqf, fkf, vbufT, obuf);

  // 5) output projection + residual x -> tbuf fp32; LN1 -> x1 (+bf16 copy)
  qt_gemm_bf16<<<dim3(CD / 128, CM / 128), 256, 0, stream>>>(
      obuf, wobT, bo, x, tbuf, nullptr, CM, CD, CD, 0, 0);
  qt_layernorm<<<CM, 256, 0, stream>>>(tbuf, g1, be1, x1, x1b);

  // 6) FFN1 (8192 x 2048 x 512) + relu -> hb bf16
  qt_gemm_bf16<<<dim3(CDFF / 128, CM / 128), 256, 0, stream>>>(
      x1b, w1bT, b1, nullptr, nullptr, hb, CM, CDFF, CD, 1, 1);

  // 7) FFN2 (8192 x 512 x 2048) + residual x1 -> tbuf; LN2 -> d_out fp32
  qt_gemm_bf16<<<dim3(CD / 128, CM / 128), 256, 0, stream>>>(
      hb, w2bT, b2, x1, tbuf, nullptr, CM, CD, CDFF, 0, 0);
  qt_layernorm<<<CM, 256, 0, stream>>>(tbuf, g2, be2, (float*)d_out, nullptr);
}
